// DecoderBlock_43370579755512
// MI455X (gfx1250) — matrix operand order repaired and checked by reading, not compiled, not run on hardware
//
#include <hip/hip_runtime.h>
#include <hip/hip_bf16.h>
#include <cstdint>

// Shapes (fixed by the reference): B=8, T=512, D=512, TS=1024, DI=1024, DS=16, DTR=32, K=4, NL=2
// GEMM precision: bf16 inputs (converted on the fly), f32 accumulate via v_wmma_f32_16x16x32_bf16.
// All GEMMs here have M % 128 == 0 and N % 64 == 0, so the non-shifted path is guard-free.

typedef __attribute__((ext_vector_type(16))) __bf16 v16bf;
typedef __attribute__((ext_vector_type(8)))  float  v8f;

__device__ __forceinline__ unsigned short f2bf(float x) {
    unsigned int u = __float_as_uint(x);
    u += 0x7FFFu + ((u >> 16) & 1u);   // round-to-nearest-even
    return (unsigned short)(u >> 16);
}

// ---------------------------------------------------------------------------
// Tiled bf16 WMMA GEMM:  C[m,n] = sum_k A[m,k] * W[n,k]   (W is (N,K) row-major,
// i.e. the standard (out_features, in_features) layout of all weights here).
// Tile: BM=128, BN=64, BK=32. 256 threads = 8 waves; wave w owns rows [16w,16w+16)
// and all 64 cols -> 4 f32 accumulators, A-fragment reused across the 4 WMMAs.
// Software-pipelined: tile k+1 is fetched into VGPRs while tile k feeds WMMA.
//   EPI: 0=none, 1=+bias, 2=softplus(x+bias)
//   ACCUM: read-modify-write C
//   SHIFT: batch-aware A-row shift (conv-transpose taps); edge rows masked to 0
//   crowScale/crowAdd: C row mapping (interleaving conv-transpose outputs)
// ---------------------------------------------------------------------------
template<int EPI, bool ACCUM, bool SHIFT>
__global__ __launch_bounds__(256)
void gemm_bf16(const float* __restrict__ A, const float* __restrict__ W,
               float* __restrict__ C, const float* __restrict__ bias,
               int Kd, int lda, int ldw, int ldc,
               int crowScale, int crowAdd, int rowsPerBatch, int rowOff)
{
    __shared__ unsigned short a_sh[128 * 40];   // 32 cols + 8 pad (keeps 16B alignment)
    __shared__ unsigned short w_sh[64 * 40];

    const int tid = threadIdx.x;
    const int wv  = tid >> 5;
    const int ln  = tid & 31;
    const int lm  = ln & 15;
    const int m0  = blockIdx.y * 128;
    const int n0  = blockIdx.x * 64;

    // ---- hoist all loop-invariant staging addresses / masks ----
    const float* aptr[4];   // per-slot A source (already includes row*lda + kq)
    float        amsk[4];   // 0 or 1 (only != 1 on SHIFT batch edges)
    int          ash [4];   // LDS ushort offset
    #pragma unroll
    for (int it = 0; it < 4; ++it) {
        int q   = tid + it * 256;          // float4 slot within 128x32 tile
        int row = q >> 3;
        int kq  = (q & 7) << 2;
        int gm  = m0 + row;
        int srow = gm;
        float msk = 1.f;
        if (SHIFT) {
            int s  = gm % rowsPerBatch;
            int s2 = s + rowOff;
            if (s2 < 0 || s2 >= rowsPerBatch) msk = 0.f;
            srow = (msk != 0.f) ? gm + rowOff : gm;   // clamp to a safe row
        }
        aptr[it] = A + (size_t)srow * lda + kq;
        amsk[it] = msk;
        ash [it] = row * 40 + kq;
    }
    const float* wptr[2];
    int          wsh [2];
    #pragma unroll
    for (int it = 0; it < 2; ++it) {
        int q   = tid + it * 256;
        int row = q >> 3;
        int kq  = (q & 7) << 2;
        wptr[it] = W + (size_t)(n0 + row) * ldw + kq;
        wsh [it] = row * 40 + kq;
    }

    v8f acc[4];
    #pragma unroll
    for (int i = 0; i < 4; ++i) {
        v8f z = {0.f, 0.f, 0.f, 0.f, 0.f, 0.f, 0.f, 0.f};
        acc[i] = z;
    }

    const int mfrag = (wv << 4) + lm;          // A row within block for this lane
    const int khalf = ln >> 4;
    const int kbA   = khalf << 3;              // A: K chunks at kbA and kbA+16
    const int kbB   = khalf << 4;              // unused: B reads kbA, kbA+16 like A

    union Frag { uint4 u4[2]; v16bf v; };

    // ---- pipeline prologue: fetch tile 0 into registers ----
    float4 ra[4], rw[2];
    #pragma unroll
    for (int it = 0; it < 4; ++it) ra[it] = *(const float4*)(aptr[it]);
    #pragma unroll
    for (int it = 0; it < 2; ++it) rw[it] = *(const float4*)(wptr[it]);

    for (int k0 = 0; k0 < Kd; k0 += 32) {
        // ---- convert current tile regs -> bf16 LDS ----
        #pragma unroll
        for (int it = 0; it < 4; ++it) {
            float4 f = ra[it];
            if (SHIFT) {
                f.x *= amsk[it]; f.y *= amsk[it];
                f.z *= amsk[it]; f.w *= amsk[it];
            }
            ushort4 h;
            h.x = f2bf(f.x); h.y = f2bf(f.y); h.z = f2bf(f.z); h.w = f2bf(f.w);
            *(ushort4*)&a_sh[ash[it]] = h;
        }
        #pragma unroll
        for (int it = 0; it < 2; ++it) {
            float4 f = rw[it];
            ushort4 h;
            h.x = f2bf(f.x); h.y = f2bf(f.y); h.z = f2bf(f.z); h.w = f2bf(f.w);
            *(ushort4*)&w_sh[wsh[it]] = h;
        }
        __syncthreads();

        // ---- fetch tile k+1 (completes behind the WMMAs below) ----
        if (k0 + 32 < Kd) {
            #pragma unroll
            for (int it = 0; it < 4; ++it) ra[it] = *(const float4*)(aptr[it] + k0 + 32);
            #pragma unroll
            for (int it = 0; it < 2; ++it) rw[it] = *(const float4*)(wptr[it] + k0 + 32);
            if (k0 + 64 < Kd) {    // prefetch the streaming A tile two steps ahead
                #pragma unroll
                for (int it = 0; it < 4; ++it)
                    __builtin_prefetch((const void*)(aptr[it] + k0 + 64), 0, 3);
            }
        }

        // ---- A fragment (ISA 7.12.2: lanes 0-15 K=[kb,kb+8)+[kb+16,kb+24)) ----
        Frag fa;
        fa.u4[0] = *(const uint4*)&a_sh[mfrag * 40 + kbA];
        fa.u4[1] = *(const uint4*)&a_sh[mfrag * 40 + kbA + 16];

        #pragma unroll
        for (int nt = 0; nt < 4; ++nt) {
            int nc = (nt << 4) + lm;
            // B fragment: same K order as A (lanes 0-15 K=[0,8)+[16,24), lanes 16-31 K=[8,16)+[24,32))
            Frag fb;
            fb.u4[0] = *(const uint4*)&w_sh[nc * 40 + kbA];
            fb.u4[1] = *(const uint4*)&w_sh[nc * 40 + kbA + 16];
            acc[nt] = __builtin_amdgcn_wmma_f32_16x16x32_bf16(
                false, fa.v, false, fb.v, (short)0, acc[nt], false, false);
        }
        __syncthreads();
    }

    // ---- epilogue (no bounds checks: M%128==0, N%64==0 for all calls) ----
    #pragma unroll
    for (int nt = 0; nt < 4; ++nt) {
        int gn = n0 + (nt << 4) + lm;
        float bv = (EPI != 0) ? bias[gn] : 0.f;
        #pragma unroll
        for (int r = 0; r < 8; ++r) {
            int row = (wv << 4) + r + ((ln >> 4) << 3);
            int gm  = m0 + row;
            float val = acc[nt][r];
            size_t co = (size_t)(crowScale * gm + crowAdd) * ldc + gn;
            if (ACCUM)    val += C[co];
            if (EPI == 1) val += bv;
            if (EPI == 2) {
                val += bv;
                val = (val > 20.f) ? val : log1pf(__expf(val));   // softplus
            }
            C[co] = val;
        }
    }
}

// ---------------------------------------------------------------------------
// up_w (Din=i, Dout=o, K) -> 4 transposed (o,i) f32 matrices for the GEMM path
// ---------------------------------------------------------------------------
__global__ void repack_upw(const float* __restrict__ up_w, float* __restrict__ U)
{
    int idx = blockIdx.x * blockDim.x + threadIdx.x;      // 4*512*512
    int k   = idx >> 18;
    int rem = idx & ((1 << 18) - 1);
    int o   = rem >> 9;
    int i   = rem & 511;
    U[((size_t)k << 18) + o * 512 + i] = up_w[((size_t)i * 512 + o) * 4 + k];
}

__global__ void copy_skip(const float* __restrict__ skip, float* __restrict__ cat)
{
    int idx = blockIdx.x * blockDim.x + threadIdx.x;      // 8192*512
    int m = idx >> 9;
    int j = idx & 511;
    cat[(size_t)m * 1024 + 512 + j] = skip[(size_t)m * 512 + j];
}

// ---------------------------------------------------------------------------
// In-place LayerNorm over rows of 512 (one block per row)
// ---------------------------------------------------------------------------
__global__ __launch_bounds__(256)
void ln_kernel(float* __restrict__ act, const float* __restrict__ g,
               const float* __restrict__ b)
{
    __shared__ float red[256];
    int row = blockIdx.x;
    float* p = act + (size_t)row * 512;
    int tid = threadIdx.x;
    float x0 = p[tid], x1 = p[tid + 256];

    red[tid] = x0 + x1;
    __syncthreads();
    for (int s = 128; s > 0; s >>= 1) { if (tid < s) red[tid] += red[tid + s]; __syncthreads(); }
    float mean = red[0] * (1.f / 512.f);
    __syncthreads();

    red[tid] = x0 * x0 + x1 * x1;
    __syncthreads();
    for (int s = 128; s > 0; s >>= 1) { if (tid < s) red[tid] += red[tid + s]; __syncthreads(); }
    float var = red[0] * (1.f / 512.f) - mean * mean;

    float rs = rsqrtf(var + 1e-5f);
    p[tid]       = (x0 - mean) * rs * g[tid]       + b[tid];
    p[tid + 256] = (x1 - mean) * rs * g[tid + 256] + b[tid + 256];
}

// ---------------------------------------------------------------------------
// Depthwise causal conv (K=4, pad 3 left) + SiLU.  xi = xz[:, 0:1024]
// ---------------------------------------------------------------------------
__global__ void conv_silu(const float* __restrict__ xz, const float* __restrict__ cw,
                          const float* __restrict__ cb, float* __restrict__ xc)
{
    int idx = blockIdx.x * blockDim.x + threadIdx.x;      // 8192*1024
    int m = idx >> 10;
    int c = idx & 1023;
    int t = m & 1023;                                     // TS = 1024
    float acc = cb[c];
    #pragma unroll
    for (int k = 0; k < 4; ++k) {
        int tt = t - 3 + k;
        if (tt >= 0) acc += xz[(size_t)(m - t + tt) * 2048 + c] * cw[c * 4 + k];
    }
    xc[(size_t)m * 1024 + c] = acc / (1.f + __expf(-acc));
}

// ---------------------------------------------------------------------------
// Selective scan: one lane per (batch, channel). 16 states in registers,
// sequential over L=1024. Fuses + D*u and * silu(z).
// ---------------------------------------------------------------------------
__global__ __launch_bounds__(256)
void scan_kernel(const float* __restrict__ dtb, const float* __restrict__ xc,
                 const float* __restrict__ xz, const float* __restrict__ xdbl,
                 const float* __restrict__ alog, const float* __restrict__ dpar,
                 float* __restrict__ y)
{
    int g = blockIdx.x * blockDim.x + threadIdx.x;        // 8192 = B*DI
    int b = g >> 10;
    int c = g & 1023;

    float Av[16], h[16];
    #pragma unroll
    for (int s = 0; s < 16; ++s) {
        Av[s] = -__expf(alog[(size_t)c * 16 + s]);
        h[s]  = 0.f;
    }
    float Dp = dpar[c];

    for (int t = 0; t < 1024; ++t) {
        size_t base = (size_t)(b << 10) + t;
        float dt = dtb[base * 1024 + c];
        float u  = xc [base * 1024 + c];
        float z  = xz [base * 2048 + 1024 + c];

        const float4* Bp = (const float4*)(xdbl + base * 64 + 32);
        const float4* Cp = (const float4*)(xdbl + base * 64 + 48);
        float Bf[16], Cf[16];
        *(float4*)&Bf[0]  = Bp[0]; *(float4*)&Bf[4]  = Bp[1];
        *(float4*)&Bf[8]  = Bp[2]; *(float4*)&Bf[12] = Bp[3];
        *(float4*)&Cf[0]  = Cp[0]; *(float4*)&Cf[4]  = Cp[1];
        *(float4*)&Cf[8]  = Cp[2]; *(float4*)&Cf[12] = Cp[3];

        float du  = dt * u;
        float acc = 0.f;
        #pragma unroll
        for (int s = 0; s < 16; ++s) {
            h[s] = __expf(dt * Av[s]) * h[s] + du * Bf[s];
            acc += h[s] * Cf[s];
        }
        acc += Dp * u;
        acc *= z / (1.f + __expf(-z));                    // * silu(z)
        y[base * 1024 + c] = acc;
    }
}

// ---------------------------------------------------------------------------
extern "C" void kernel_launch(void* const* d_in, const int* in_sizes, int n_in,
                              void* d_out, int out_size, void* d_ws, size_t ws_size,
                              hipStream_t stream)
{
    (void)in_sizes; (void)n_in; (void)out_size; (void)ws_size;
    const float* x         = (const float*)d_in[0];
    const float* skip      = (const float*)d_in[1];
    const float* up_w      = (const float*)d_in[2];
    const float* up_b      = (const float*)d_in[3];
    const float* merge_w   = (const float*)d_in[4];
    const float* merge_b   = (const float*)d_in[5];
    const float* ln_w      = (const float*)d_in[6];
    const float* ln_b      = (const float*)d_in[7];
    const float* in_proj_w = (const float*)d_in[8];
    const float* conv_w    = (const float*)d_in[9];
    const float* conv_b    = (const float*)d_in[10];
    const float* x_proj_w  = (const float*)d_in[11];
    const float* dt_proj_w = (const float*)d_in[12];
    const float* dt_proj_b = (const float*)d_in[13];
    const float* A_log     = (const float*)d_in[14];
    const float* D_param   = (const float*)d_in[15];
    const float* out_proj_w= (const float*)d_in[16];

    // workspace layout (floats)
    float* ws   = (float*)d_ws;
    float* ACT  = ws;                        // 8192*512   layer input / LN output
    float* CAT  = ACT  + (size_t)8192 * 512; // 8192*1024  cat buffer, reused as y
    float* XZ   = CAT  + (size_t)8192 * 1024;// 8192*2048
    float* XC   = XZ   + (size_t)8192 * 2048;// 8192*1024
    float* XDBL = XC   + (size_t)8192 * 1024;// 8192*64
    float* DTB  = XDBL + (size_t)8192 * 64;  // 8192*1024
    float* UWT  = DTB  + (size_t)8192 * 1024;// 4*512*512  transposed up_w taps

    dim3 blk(256);

    repack_upw<<<dim3(4 * 512 * 512 / 256), blk, 0, stream>>>(up_w, UWT);
    copy_skip <<<dim3(8192 * 512 / 256),    blk, 0, stream>>>(skip, CAT);

    // Transposed conv as 4 per-parity GEMMs writing interleaved rows of CAT[:, :512]:
    //   even t=2s:  X[s]@U1 + X[s-1]@U3 + up_b
    //   odd  t=2s+1:X[s]@U2 + X[s+1]@U0 + up_b
    gemm_bf16<1, false, false><<<dim3(8, 32), blk, 0, stream>>>(x, UWT + 1 * 262144, CAT, up_b,
        512, 512, 512, 1024, 2, 0, 512, 0);
    gemm_bf16<0, true,  true ><<<dim3(8, 32), blk, 0, stream>>>(x, UWT + 3 * 262144, CAT, nullptr,
        512, 512, 512, 1024, 2, 0, 512, -1);
    gemm_bf16<1, false, false><<<dim3(8, 32), blk, 0, stream>>>(x, UWT + 2 * 262144, CAT, up_b,
        512, 512, 512, 1024, 2, 1, 512, 0);
    gemm_bf16<0, true,  true ><<<dim3(8, 32), blk, 0, stream>>>(x, UWT + 0,          CAT, nullptr,
        512, 512, 512, 1024, 2, 1, 512, 1);

    // merge GEMM + LayerNorm -> ACT
    gemm_bf16<1, false, false><<<dim3(8, 64), blk, 0, stream>>>(CAT, merge_w, ACT, merge_b,
        1024, 1024, 1024, 512, 1, 0, 8192, 0);
    ln_kernel<<<dim3(8192), blk, 0, stream>>>(ACT, ln_w, ln_b);

    for (int i = 0; i < 2; ++i) {
        // xz = act @ in_proj_w[i]^T   (8192 x 2048)
        gemm_bf16<0, false, false><<<dim3(32, 64), blk, 0, stream>>>(ACT,
            in_proj_w + (size_t)i * 2048 * 512, XZ, nullptr,
            512, 512, 512, 2048, 1, 0, 8192, 0);
        // depthwise conv + silu -> xc
        conv_silu<<<dim3(8192 * 1024 / 256), blk, 0, stream>>>(XZ,
            conv_w + (size_t)i * 1024 * 4, conv_b + (size_t)i * 1024, XC);
        // xdbl = xc @ x_proj_w[i]^T   (8192 x 64)
        gemm_bf16<0, false, false><<<dim3(1, 64), blk, 0, stream>>>(XC,
            x_proj_w + (size_t)i * 64 * 1024, XDBL, nullptr,
            1024, 1024, 1024, 64, 1, 0, 8192, 0);
        // dt = softplus(xdbl[:, :32] @ dt_proj_w[i]^T + dt_proj_b[i])
        gemm_bf16<2, false, false><<<dim3(16, 64), blk, 0, stream>>>(XDBL,
            dt_proj_w + (size_t)i * 1024 * 32, DTB, dt_proj_b + (size_t)i * 1024,
            32, 64, 32, 1024, 1, 0, 8192, 0);
        // selective scan (+D*u, *silu(z)) -> y (reuse CAT)
        scan_kernel<<<dim3(32), blk, 0, stream>>>(DTB, XC, XZ, XDBL,
            A_log + (size_t)i * 1024 * 16, D_param + (size_t)i * 1024, CAT);
        // out = y @ out_proj_w[i]^T
        float* OUTP = (i == 0) ? ACT : (float*)d_out;
        gemm_bf16<0, false, false><<<dim3(8, 64), blk, 0, stream>>>(CAT,
            out_proj_w + (size_t)i * 512 * 1024, OUTP, nullptr,
            1024, 1024, 1024, 512, 1, 0, 8192, 0);
    }
}
